// LocalSlidingWindowAttention_55284819034488
// MI455X (gfx1250) — matrix operand order repaired and checked by reading, not compiled, not run on hardware
//
#include <hip/hip_runtime.h>
#include <stdint.h>

// ---------- types ----------
typedef __attribute__((ext_vector_type(16))) __bf16        v16bf;
typedef __attribute__((ext_vector_type(8)))  float         v8f;
typedef __attribute__((ext_vector_type(4)))  unsigned int  u32x4;

#define BB   2
#define TT   2048
#define DD   1024
#define HH   16
#define HD   64
#define WIN  64           // query i attends keys j with 0 <= j - i <= 64
#define PLANE (BB * HH * TT * HD)   // 4,194,304 elements per Q/K/V plane

union Frag16 {
  v16bf          v;
  u32x4          q[2];
  unsigned short u[16];
};

__device__ __forceinline__ unsigned short f2bf(float x) {
  unsigned int u = __float_as_uint(x);
  unsigned int r = u + 0x7FFFu + ((u >> 16) & 1u);   // round-to-nearest-even
  return (unsigned short)(r >> 16);
}

// ---------- CDNA5 async global->LDS copies (ASYNCcnt-tracked) ----------
__device__ __forceinline__ void async_ld_b128(unsigned int lds_off,
                                              const unsigned short* gptr) {
  asm volatile("global_load_async_to_lds_b128 %0, %1, off"
               :: "v"(lds_off), "v"(gptr) : "memory");
}
__device__ __forceinline__ void wait_async3() {
  asm volatile("s_wait_asynccnt 0x3" ::: "memory");
}
__device__ __forceinline__ void wait_async0() {
  asm volatile("s_wait_asynccnt 0x0" ::: "memory");
}
__device__ __forceinline__ unsigned int lds_off_of(const void* p) {
  return (unsigned int)(uintptr_t)p;   // LDS offset is the low 32 bits of the address
}

// ---------- fp32 -> bf16 conversion ----------
__global__ void cvt_f32_bf16(const float* __restrict__ in,
                             unsigned short* __restrict__ out, int n) {
  int i = blockIdx.x * blockDim.x + threadIdx.x;
  if (i < n) out[i] = f2bf(in[i]);
}

// ---------- bf16 WMMA GEMM: out[m,n] = sum_k A[m,k]*W[n,k] + bias[n] ----------
// block: 256 threads (8 waves), tile 128(M) x 64(N), K-step 32.
// Each wave: 1 A-fragment reused across 4 N-subtiles -> 4 WMMAs / K-step.
// Double-buffered LDS fed by async global->LDS loads (3 ops/thread/stage).
// mode 0: write fp32 row-major M x N | mode 1: bf16 scatter to Q/K/V planes
__launch_bounds__(256)
__global__ void gemm_bf16_wmma(const unsigned short* __restrict__ A,
                               const unsigned short* __restrict__ W,
                               const float* __restrict__ bias,
                               float* __restrict__ outF,
                               unsigned short* __restrict__ outQKV,
                               int M, int N, int K, int mode) {
  __shared__ __align__(16) unsigned short As[2][128][48];
  __shared__ __align__(16) unsigned short Bs[2][64][48];

  const int tid  = threadIdx.x;
  const int wave = tid >> 5;
  const int lane = tid & 31;
  const int half = lane >> 4;
  const int l16  = lane & 15;

  const int m0 = blockIdx.x * 128;
  const int n0 = blockIdx.y * 64;

  const int a_row = tid >> 1, a_h = tid & 1;   // A: 32 B per thread (2 x b128)
  const int b_n   = tid >> 2, b_c = tid & 3;   // B: 16 B per thread (1 x b128)

  auto issue_stage = [&](int k0, int nb) {
    const unsigned short* asrc = A + (size_t)(m0 + a_row) * K + k0 + a_h * 16;
    unsigned int al = lds_off_of(&As[nb][a_row][a_h * 16]);
    async_ld_b128(al,      asrc);
    async_ld_b128(al + 16, asrc + 8);
    const unsigned short* bsrc = W + (size_t)(n0 + b_n) * K + k0 + b_c * 8;
    async_ld_b128(lds_off_of(&Bs[nb][b_n][b_c * 8]), bsrc);
  };

  v8f acc[4] = {{}, {}, {}, {}};

  auto compute = [&](int cur) {
    Frag16 af;                               // A frag: row = wave*16 + l16
    af.q[0] = *(const u32x4*)&As[cur][wave * 16 + l16][half * 8];
    af.q[1] = *(const u32x4*)&As[cur][wave * 16 + l16][16 + half * 8];
    Frag16 bf[4];                            // preload all B frags first
    #pragma unroll
    for (int ns = 0; ns < 4; ++ns) {
      bf[ns].q[0] = *(const u32x4*)&Bs[cur][ns * 16 + l16][half * 8];
      bf[ns].q[1] = *(const u32x4*)&Bs[cur][ns * 16 + l16][16 + half * 8];
    }
    #pragma unroll
    for (int ns = 0; ns < 4; ++ns)
      acc[ns] = __builtin_amdgcn_wmma_f32_16x16x32_bf16(
          false, af.v, false, bf[ns].v, (short)0, acc[ns], false, false);
  };

  const int NK = K >> 5;
  issue_stage(0, 0);

  int it = 0;
  for (; it < NK - 1; ++it) {
    issue_stage((it + 1) << 5, (it & 1) ^ 1);
    wait_async3();                 // this thread's stage `it` ops complete
    __syncthreads();               // everyone's stage `it` data visible
    compute(it & 1);
    __syncthreads();               // done reading before buffer is refilled
  }
  wait_async0();                   // peeled final stage
  __syncthreads();
  compute(it & 1);

  // epilogue: D layout -> M = v + 8*half, N = l16
  #pragma unroll
  for (int ns = 0; ns < 4; ++ns) {
    #pragma unroll
    for (int v = 0; v < 8; ++v) {
      int m = m0 + wave * 16 + v + half * 8;
      int n = n0 + ns * 16 + l16;
      float val = acc[ns][v] + bias[n];
      if (mode == 0) {
        outF[(size_t)m * N + n] = val;
      } else {
        int b = m >> 11, t = m & (TT - 1);
        int three = n >> 10, h = (n >> 6) & 15, dd = n & 63;
        size_t idx = (size_t)three * PLANE +
                     (((size_t)(b * HH + h)) * TT + t) * HD + dd;
        outQKV[idx] = f2bf(val);
      }
    }
  }
}

// ---------- V transpose: per (b,h), (T,hd) -> (hd,T) ----------
// block 256 threads, 64(keys) x 64(dims) tile through LDS; both global sides
// coalesced; the scalar gather runs once per element (attention reuses 2.5x).
__launch_bounds__(256)
__global__ void transpose_v(const unsigned short* __restrict__ V,
                            unsigned short* __restrict__ Vt) {
  __shared__ __align__(16) unsigned short Tl[64][72];

  const int tid  = threadIdx.x;
  const int bh   = blockIdx.x;
  const int key0 = blockIdx.y * 64;
  const size_t vbase  = (size_t)bh * TT * HD;
  const size_t vtbase = (size_t)bh * HD * TT;

  // stage 64 keys x 64 dims (async, coalesced)
  #pragma unroll
  for (int i = 0; i < 2; ++i) {
    int c  = tid + i * 256;         // 512 chunks of 16 B
    int k  = c >> 3;
    int ch = c & 7;
    async_ld_b128(lds_off_of(&Tl[k][ch * 8]),
                  V + vbase + (size_t)(key0 + k) * HD + ch * 8);
  }
  wait_async0();
  __syncthreads();

  // write transposed: fixed dim, contiguous keys
  const int dim = tid >> 2;
  const int j   = tid & 3;
  Frag16 vv;
  #pragma unroll
  for (int e = 0; e < 16; ++e) vv.u[e] = Tl[j * 16 + e][dim];
  unsigned short* dst = Vt + vtbase + (size_t)dim * TT + key0 + j * 16;
  *(u32x4*)(dst + 0) = vv.q[0];
  *(u32x4*)(dst + 8) = vv.q[1];
}

// ---------- sliding-window attention ----------
// block: 128 threads (4 waves), 64 queries of one (b,h); wave -> 16 queries.
// V^T slab (64 dims x 160 keys) async-staged to LDS, overlapped with
// QK^T + softmax; every V B-fragment is then two contiguous ds_load_b128.
__launch_bounds__(128)
__global__ void attn_wmma(const unsigned short* __restrict__ Qg,
                          const unsigned short* __restrict__ Kg,
                          const unsigned short* __restrict__ Vt,
                          unsigned short* __restrict__ Og) {
  __shared__ __align__(16) unsigned short Vtlds[64][168];  // dims x keys (pad)
  __shared__ __align__(16) unsigned short Plds[4][16][96];

  const int wave = threadIdx.x >> 5;
  const int lane = threadIdx.x & 31;
  const int half = lane >> 4;
  const int l16  = lane & 15;

  const int bh  = blockIdx.x;                // b*H + h
  const int q0b = blockIdx.y * 64;
  const int q0  = q0b + wave * 16;
  const size_t base   = (size_t)bh * TT * HD;
  const size_t vtbase = (size_t)bh * HD * TT;
  const int b = bh / HH, h = bh % HH;

  // ---- issue async V^T staging: 64 dims x keys [q0b, q0b+159] ----
  #pragma unroll
  for (int i = 0; i < 10; ++i) {
    int c   = threadIdx.x + i * 128;         // 1280 chunks of 16 B
    int row = c / 20;                        // dim
    int off = (c % 20) * 8;                  // key offset within slab
    int ks  = q0b + off;
    if (ks > TT - 8) ks = TT - 8;            // clamp; masked keys are P==0
    async_ld_b128(lds_off_of(&Vtlds[row][off]),
                  Vt + vtbase + (size_t)row * TT + ks);
  }

  // ---- Q as A-matrix 16x64 -> two K=32 fragments ----
  Frag16 qa[2];
  {
    const unsigned short* qp = Qg + base + (size_t)(q0 + l16) * HD;
    #pragma unroll
    for (int f = 0; f < 2; ++f) {
      qa[f].q[0] = *(const u32x4*)(qp + f * 32 + half * 8);
      qa[f].q[1] = *(const u32x4*)(qp + f * 32 + 16 + half * 8);
    }
  }

  const float scale = 0.125f;                // hd^-0.5
  v8f s[6];

  #pragma unroll
  for (int kt = 0; kt < 6; ++kt) {           // 96-key window (80 used + mask)
    int kb  = q0 + kt * 16;
    int key = kb + l16;
    int keyc = key < TT ? key : TT - 1;
    const unsigned short* kp = Kg + base + (size_t)keyc * HD;
    Frag16 kf0, kf1;                         // K^T as B-matrix: table-order hd
    kf0.q[0] = *(const u32x4*)(kp + half * 8);
    kf0.q[1] = *(const u32x4*)(kp + 16 + half * 8);
    kf1.q[0] = *(const u32x4*)(kp + 32 + half * 8);
    kf1.q[1] = *(const u32x4*)(kp + 48 + half * 8);

    v8f c = {};
    c = __builtin_amdgcn_wmma_f32_16x16x32_bf16(false, qa[0].v, false, kf0.v,
                                                (short)0, c, false, false);
    c = __builtin_amdgcn_wmma_f32_16x16x32_bf16(false, qa[1].v, false, kf1.v,
                                                (short)0, c, false, false);
    #pragma unroll
    for (int v = 0; v < 8; ++v) {
      int qi = q0 + v + half * 8;
      int kj = kb + l16;
      bool valid = (kj >= qi) && (kj - qi <= WIN) && (kj < TT);
      c[v] = valid ? c[v] * scale : -__builtin_inff();
    }
    s[kt] = c;
  }

  // ---- masked softmax: reductions across the 16-lane half-groups ----
  float rsum[8];
  #pragma unroll
  for (int v = 0; v < 8; ++v) {
    float m = s[0][v];
    #pragma unroll
    for (int kt = 1; kt < 6; ++kt) m = fmaxf(m, s[kt][v]);
    #pragma unroll
    for (int off = 1; off < 16; off <<= 1) m = fmaxf(m, __shfl_xor(m, off, 32));
    float sum = 0.f;
    #pragma unroll
    for (int kt = 0; kt < 6; ++kt) {
      float e = __expf(s[kt][v] - m);        // masked entries: exp(-inf) = 0
      s[kt][v] = e;
      sum += e;
    }
    #pragma unroll
    for (int off = 1; off < 16; off <<= 1) sum += __shfl_xor(sum, off, 32);
    rsum[v] = sum;                           // >= 1 (diagonal always valid)
  }

  // ---- P -> LDS (bf16): D-fragment layout re-shaped to A-fragment layout ----
  #pragma unroll
  for (int kt = 0; kt < 6; ++kt)
    #pragma unroll
    for (int v = 0; v < 8; ++v)
      Plds[wave][v + half * 8][kt * 16 + l16] = f2bf(s[kt][v]);

  wait_async0();                             // V^T slab landed (this thread)
  __syncthreads();                           // all threads' V^T + P visible

  Frag16 pa[3];
  #pragma unroll
  for (int f = 0; f < 3; ++f) {
    pa[f].q[0] = *(const u32x4*)&Plds[wave][l16][f * 32 + half * 8];
    pa[f].q[1] = *(const u32x4*)&Plds[wave][l16][f * 32 + 16 + half * 8];
  }

  // ---- O = P (16x96) x V (96x64); V B-fragments: contiguous keys per dim ----
  v8f o[4] = {{}, {}, {}, {}};
  #pragma unroll
  for (int f = 0; f < 3; ++f) {
    #pragma unroll
    for (int nt = 0; nt < 4; ++nt) {
      Frag16 vb;
      int dim = nt * 16 + l16;
      int kloc = wave * 16 + f * 32 + half * 8;    // local key base
      vb.q[0] = *(const u32x4*)&Vtlds[dim][kloc];
      vb.q[1] = *(const u32x4*)&Vtlds[dim][kloc + 16];
      o[nt] = __builtin_amdgcn_wmma_f32_16x16x32_bf16(false, pa[f].v, false, vb.v,
                                                      (short)0, o[nt], false, false);
    }
  }

  // ---- normalize and store as (B,T,D) bf16 ----
  #pragma unroll
  for (int nt = 0; nt < 4; ++nt)
    #pragma unroll
    for (int v = 0; v < 8; ++v) {
      int t = q0 + v + half * 8;
      float val = o[nt][v] / rsum[v];
      size_t idx = ((size_t)b * TT + t) * DD + h * HD + nt * 16 + l16;
      Og[idx] = f2bf(val);
    }
}

// ---------- launcher ----------
extern "C" void kernel_launch(void* const* d_in, const int* in_sizes, int n_in,
                              void* d_out, int out_size, void* d_ws, size_t ws_size,
                              hipStream_t stream) {
  const float* x     = (const float*)d_in[0];
  const float* w_qkv = (const float*)d_in[1];
  const float* b_qkv = (const float*)d_in[2];
  const float* w_out = (const float*)d_in[3];
  const float* b_out = (const float*)d_in[4];

  char* ws = (char*)d_ws;
  unsigned short* xb    = (unsigned short*)(ws + 0);         //  8 MB
  unsigned short* wqkvb = (unsigned short*)(ws + 8388608);   //  6 MB
  unsigned short* woutb = (unsigned short*)(ws + 14680064);  //  2 MB
  unsigned short* qkvb  = (unsigned short*)(ws + 16777216);  // 24 MB (3 planes)
  unsigned short* attnb = (unsigned short*)(ws + 41943040);  //  8 MB
  unsigned short* vtb   = (unsigned short*)(ws + 50331648);  //  8 MB (V^T)

  const int nx  = BB * TT * DD;   // 4,194,304
  const int nwq = 3 * DD * DD;    // 3,145,728
  const int nwo = DD * DD;        // 1,048,576

  cvt_f32_bf16<<<(nx  + 255) / 256, 256, 0, stream>>>(x,     xb,    nx);
  cvt_f32_bf16<<<(nwq + 255) / 256, 256, 0, stream>>>(w_qkv, wqkvb, nwq);
  cvt_f32_bf16<<<(nwo + 255) / 256, 256, 0, stream>>>(w_out, woutb, nwo);

  // QKV projection: M=4096, N=3072, K=1024 -> scatter into Q/K/V planes
  {
    dim3 grid(4096 / 128, 3072 / 64);
    gemm_bf16_wmma<<<grid, 256, 0, stream>>>(xb, wqkvb, b_qkv,
                                             nullptr, qkvb,
                                             4096, 3072, 1024, /*mode=*/1);
  }

  // V -> V^T per (b,h)
  {
    dim3 grid(BB * HH, TT / 64);
    transpose_v<<<grid, 256, 0, stream>>>(qkvb + 2 * PLANE, vtb);
  }

  // windowed attention
  {
    dim3 grid(BB * HH, TT / 64);
    attn_wmma<<<grid, 128, 0, stream>>>(qkvb, qkvb + PLANE, vtb, attnb);
  }

  // output projection: M=4096, N=1024, K=1024 -> fp32 d_out
  {
    dim3 grid(4096 / 128, 1024 / 64);
    gemm_bf16_wmma<<<grid, 256, 0, stream>>>(attnb, woutb, b_out,
                                             (float*)d_out, nullptr,
                                             4096, 1024, 1024, /*mode=*/0);
  }
}
